// MultiHeadSelfAttention_9028021256338
// MI455X (gfx1250) — compile-verified, matrix operand order checked by reading, not run on hardware
//
#include <hip/hip_runtime.h>

typedef __attribute__((ext_vector_type(16))) _Float16 v16h;
typedef __attribute__((ext_vector_type(8)))  _Float16 v8h;
typedef __attribute__((ext_vector_type(8)))  float    v8f;

#define D_MODEL  1024
#define N_HEADS  16
#define HEAD_DIM 64
#define BATCH    4
#define SEQ      2048
#define SEQP     (SEQ + 32)   // padded K/V length so diagonal-block overfetch is in-bounds

// ---------------------------------------------------------------------------
// CDNA5 primitives
// ---------------------------------------------------------------------------
__device__ __forceinline__ v8f wmma_f16(v16h a, v16h b, v8f c) {
  // 8 args: (neg_a, A, neg_b, B, c_mod, C, reuse_a, reuse_b)
  return __builtin_amdgcn_wmma_f32_16x16x32_f16(false, a, false, b, (short)0, c,
                                                false, false);
}

// Work-group split barrier WITHOUT the compiler's conservative vmem flush:
// hardware needs no counter waits at a barrier; we emit only the counter waits
// that are semantically required (asynccnt / dscnt) ourselves.
__device__ __forceinline__ void wg_barrier() {
  asm volatile("s_barrier_signal -1\n\ts_barrier_wait -1" ::: "memory");
}

// Async global->LDS stage of 16 bytes per lane (ASYNCcnt-tracked, no VGPRs).
__device__ __forceinline__ void async_stage_b128(unsigned lds_off,
                                                 const _Float16* gptr) {
  asm volatile("global_load_async_to_lds_b128 %0, %1, off"
               :: "v"(lds_off), "v"(gptr) : "memory");
}

// Load a 16x32 f16 fragment (A layout; identical per-lane layout used for B with
// "row" = output column, since our B matrices are stored [N,K] row-major).
// ISA 16-bit A 16x32 layout: lane L holds row M=L%16; K chunks of 8:
//   lanes 0-15 : K = 0..7   and 16..23
//   lanes 16-31: K = 8..15  and 24..31
__device__ __forceinline__ v16h load_frag(const _Float16* __restrict__ base,
                                          int ld, int row, int k0, int lane) {
  const _Float16* p = base + (long)(row + (lane & 15)) * ld + k0 + ((lane >> 4) << 3);
  v8h lo = *(const v8h*)p;
  v8h hi = *(const v8h*)(p + 16);
  v16h r;
#pragma unroll
  for (int i = 0; i < 8; ++i) { r[i] = lo[i]; r[i + 8] = hi[i]; }
  return r;
}

// Same fragment gather from an LDS tile with fixed row stride 32 halfs.
__device__ __forceinline__ v16h lds_frag(const _Float16* pr) {
  v8h lo = *(const v8h*)pr;
  v8h hi = *(const v8h*)(pr + 16);
  v16h r;
#pragma unroll
  for (int i = 0; i < 8; ++i) { r[i] = lo[i]; r[i + 8] = hi[i]; }
  return r;
}

// width-16 cross-lane reductions (rows of a C-layout tile live across 16 lanes)
__device__ __forceinline__ float rmax16(float v) {
  v = fmaxf(v, __shfl_xor(v, 1, 16));
  v = fmaxf(v, __shfl_xor(v, 2, 16));
  v = fmaxf(v, __shfl_xor(v, 4, 16));
  v = fmaxf(v, __shfl_xor(v, 8, 16));
  return v;
}
__device__ __forceinline__ float rsum16(float v) {
  v += __shfl_xor(v, 1, 16);
  v += __shfl_xor(v, 2, 16);
  v += __shfl_xor(v, 4, 16);
  v += __shfl_xor(v, 8, 16);
  return v;
}

// ---------------------------------------------------------------------------
// f32 -> f16 convert
// ---------------------------------------------------------------------------
__global__ void cvt_f32_f16(const float* __restrict__ in, _Float16* __restrict__ out,
                            long n) {
  long i = (long)blockIdx.x * blockDim.x + threadIdx.x;
  long stride = (long)gridDim.x * blockDim.x;
  for (; i < n; i += stride) out[i] = (_Float16)in[i];
}

// ---------------------------------------------------------------------------
// GEMM: C[M,N] = A[M,K](f16) @ Bw[N,K]^T(f16) + bias[N]
// 8 waves/block; block tile 256(M) x 64(N); each wave owns a 32x64 strip
// (two 16-row M-tiles -> 8 WMMAs per k-step, every LDS B fragment reused x2).
// The 64x32 B tile (shared by all 8 waves) flows through a 4-deep LDS pipeline
// with distance-2 async-to-LDS prefetch (ASYNCcnt, power-of-2 buffer index);
// waves register-double-buffer their private A fragments. All four B fragments
// are gathered into distinct registers BEFORE the WMMA burst so the ds_loads
// issue as one clause with counted dscnt waits (no per-pair WAR bubbles).
// Split barriers are raw s_barrier_signal/wait so private loads stay in flight.
// ---------------------------------------------------------------------------
__global__ __launch_bounds__(256) void gemm_wmma(
    const _Float16* __restrict__ A, const _Float16* __restrict__ Bw,
    const float* __restrict__ bias, float* __restrict__ Cf,
    _Float16* __restrict__ Ch, int M, int N, int K) {
  __shared__ __attribute__((aligned(16))) _Float16 ldsB[4][64][32];  // 16 KB

  int tid  = threadIdx.x;
  int lane = tid & 31;
  int wave = tid >> 5;
  int m0 = blockIdx.y * 256 + wave * 32;   // two m-tiles: m0 and m0+16
  int n0 = blockIdx.x * 64;

  // cooperative async B staging: thread t -> local row n = t>>2, k-chunk (t&3)*8
  int bn = tid >> 2;
  int bk = (tid & 3) * 8;
  const _Float16* gB = Bw + (long)(n0 + bn) * K + bk;
  // flat->LDS: hardware uses addr[31:0] as the LDS byte address (ISA 10.2)
  unsigned ldsOff[4];
#pragma unroll
  for (int p = 0; p < 4; ++p) ldsOff[p] = (unsigned)(size_t)&ldsB[p][bn][bk];

  int nIter = K / 32;
  // pre-issue k-steps 0 and 1
  async_stage_b128(ldsOff[0], gB);
  if (nIter > 1) async_stage_b128(ldsOff[1], gB + 32);

  v8f acc[8];  // [0..3] m-tile0 x n-tiles, [4..7] m-tile1 x n-tiles
#pragma unroll
  for (int j = 0; j < 8; ++j)
#pragma unroll
    for (int r = 0; r < 8; ++r) acc[j][r] = 0.f;

  // register prefetch of first A fragments (both m-tiles)
  v16h a0_cur = load_frag(A, K, m0, 0, lane);
  v16h a1_cur = load_frag(A, K, m0 + 16, 0, lane);

  for (int it = 0; it < nIter; ++it) {
    int cur = it & 3;

    // prefetch next A fragments early; they may stay in flight across barriers
    v16h a0_nxt = a0_cur, a1_nxt = a1_cur;
    if (it + 1 < nIter) {
      a0_nxt = load_frag(A, K, m0, (it + 1) * 32, lane);
      a1_nxt = load_frag(A, K, m0 + 16, (it + 1) * 32, lane);
    }

    // distance-2 async prefetch, then wait so the oldest outstanding (current
    // tile) is resident; async-to-LDS ops complete in order.
    if (it + 2 < nIter) {
      async_stage_b128(ldsOff[(it + 2) & 3], gB + (long)(it + 2) * 32);
      asm volatile("s_wait_asynccnt 0x2" ::: "memory");
    } else if (it + 1 < nIter) {
      asm volatile("s_wait_asynccnt 0x1" ::: "memory");
    } else {
      asm volatile("s_wait_asynccnt 0x0" ::: "memory");
    }
    wg_barrier();  // current tile visible to all waves

    // gather ALL four B fragments into distinct registers (one ds clause,
    // counted waits), then burst the 8 WMMAs (each fragment reused x2).
    const _Float16* base = &ldsB[cur][0][0];
    v16h bf0 = lds_frag(base + (0 * 16 + (lane & 15)) * 32 + ((lane >> 4) << 3));
    v16h bf1 = lds_frag(base + (1 * 16 + (lane & 15)) * 32 + ((lane >> 4) << 3));
    v16h bf2 = lds_frag(base + (2 * 16 + (lane & 15)) * 32 + ((lane >> 4) << 3));
    v16h bf3 = lds_frag(base + (3 * 16 + (lane & 15)) * 32 + ((lane >> 4) << 3));

    acc[0] = wmma_f16(a0_cur, bf0, acc[0]);
    acc[4] = wmma_f16(a1_cur, bf0, acc[4]);
    acc[1] = wmma_f16(a0_cur, bf1, acc[1]);
    acc[5] = wmma_f16(a1_cur, bf1, acc[5]);
    acc[2] = wmma_f16(a0_cur, bf2, acc[2]);
    acc[6] = wmma_f16(a1_cur, bf2, acc[6]);
    acc[3] = wmma_f16(a0_cur, bf3, acc[3]);
    acc[7] = wmma_f16(a1_cur, bf3, acc[7]);

    asm volatile("s_wait_dscnt 0x0" ::: "memory");  // reads landed in VGPRs
    wg_barrier();  // all waves done reading -> buffer may be overwritten

    a0_cur = a0_nxt;
    a1_cur = a1_nxt;
  }

  int rowhalf = (lane >> 4) << 3;  // C layout: VGPR r holds M = r (+8 for lanes 16-31)
#pragma unroll
  for (int mt = 0; mt < 2; ++mt) {
#pragma unroll
    for (int j = 0; j < 4; ++j) {
      int n = n0 + j * 16 + (lane & 15);
      float bv = bias[n];
#pragma unroll
      for (int r = 0; r < 8; ++r) {
        long idx = (long)(m0 + mt * 16 + r + rowhalf) * N + n;
        float v = acc[mt * 4 + j][r] + bv;
        if (Ch) Ch[idx] = (_Float16)v;
        else    Cf[idx] = v;
      }
    }
  }
}

// ---------------------------------------------------------------------------
// RoPE + head pack: qkv[B,S,3072](f16) -> qh[bh][S][64], kh[bh][SEQP][64],
// vT[bh][64][SEQP]; pad region (s >= SEQ) zeroed for k and v.
// ---------------------------------------------------------------------------
__global__ void rope_pack(const _Float16* __restrict__ qkv,
                          const int* __restrict__ positions,
                          _Float16* __restrict__ qh, _Float16* __restrict__ kh,
                          _Float16* __restrict__ vT) {
  long tid = (long)blockIdx.x * blockDim.x + threadIdx.x;
  int i = (int)(tid & 31); tid >>= 5;
  int s = (int)(tid % SEQP); tid /= SEQP;
  int bh = (int)tid;
  if (bh >= BATCH * N_HEADS) return;
  int b = bh >> 4, h = bh & 15;

  long kidx = ((long)bh * SEQP + s) * HEAD_DIM + 2 * i;
  long vidx0 = ((long)bh * HEAD_DIM + 2 * i) * SEQP + s;
  long vidx1 = vidx0 + SEQP;

  if (s >= SEQ) {  // zero pad so masked-out tiles contribute exact zeros
    kh[kidx] = (_Float16)0.f; kh[kidx + 1] = (_Float16)0.f;
    vT[vidx0] = (_Float16)0.f; vT[vidx1] = (_Float16)0.f;
    return;
  }

  long in_row = ((long)b * SEQ + s) * (3 * D_MODEL);
  int d0 = h * HEAD_DIM + 2 * i;
  float q1 = (float)qkv[in_row + d0];
  float q2 = (float)qkv[in_row + d0 + 1];
  float k1 = (float)qkv[in_row + D_MODEL + d0];
  float k2 = (float)qkv[in_row + D_MODEL + d0 + 1];
  float v1 = (float)qkv[in_row + 2 * D_MODEL + d0];
  float v2 = (float)qkv[in_row + 2 * D_MODEL + d0 + 1];

  float pos = (float)positions[b * SEQ + s];
  float inv_freq = __powf(10000.f, -((float)(2 * i) / (float)HEAD_DIM));
  float ang = pos * inv_freq;
  float sn, cs;
  __sincosf(ang, &sn, &cs);

  long qidx = ((long)bh * SEQ + s) * HEAD_DIM + 2 * i;
  qh[qidx]     = (_Float16)(q1 * cs - q2 * sn);
  qh[qidx + 1] = (_Float16)(q1 * sn + q2 * cs);
  kh[kidx]     = (_Float16)(k1 * cs - k2 * sn);
  kh[kidx + 1] = (_Float16)(k1 * sn + k2 * cs);
  vT[vidx0] = (_Float16)v1;
  vT[vidx1] = (_Float16)v2;
}

// ---------------------------------------------------------------------------
// Flash attention, causal. Grid: (B*H, SEQ/128). 8 waves/block, each wave owns
// a 16-row Q tile, loops k-blocks of 32, online softmax, WMMA for QK^T and PV.
// K fragments register-double-buffered across k-blocks; V fragments issued at
// the top of the block so their latency hides behind the softmax VALU chain.
// P transposed C-layout -> A-layout via per-wave LDS tile (+ s_wait_dscnt).
// No block barriers: per-wave trip counts differ under the causal mask.
// ---------------------------------------------------------------------------
__global__ __launch_bounds__(256) void attn_flash(
    const _Float16* __restrict__ qh, const _Float16* __restrict__ kh,
    const _Float16* __restrict__ vT, _Float16* __restrict__ hbuf) {
  __shared__ __attribute__((aligned(16))) _Float16 lds[8][16][32];

  int lane = threadIdx.x & 31;
  int wave = threadIdx.x >> 5;
  int bh = blockIdx.x;
  int b = bh >> 4, h = bh & 15;
  int q0 = blockIdx.y * 128 + wave * 16;

  const _Float16* Q = qh + (long)bh * SEQ * HEAD_DIM;
  const _Float16* K = kh + (long)bh * SEQP * HEAD_DIM;
  const _Float16* V = vT + (long)bh * HEAD_DIM * SEQP;

  v16h aq0 = load_frag(Q, HEAD_DIM, q0, 0, lane);
  v16h aq1 = load_frag(Q, HEAD_DIM, q0, 32, lane);

  v8f o[4];
  float mrow[8], lrow[8];
#pragma unroll
  for (int j = 0; j < 4; ++j)
#pragma unroll
    for (int r = 0; r < 8; ++r) o[j][r] = 0.f;
#pragma unroll
  for (int r = 0; r < 8; ++r) { mrow[r] = -1e30f; lrow[r] = 0.f; }

  int rowhalf = (lane >> 4) << 3;
  int kend = q0 + 16;  // causal: need k <= q, q_max = q0+15

  // prefetch first k-block's K fragments
  v16h kf0 = load_frag(K, HEAD_DIM, 0, 0, lane);
  v16h kf1 = load_frag(K, HEAD_DIM, 16, 0, lane);
  v16h kf2 = load_frag(K, HEAD_DIM, 0, 32, lane);
  v16h kf3 = load_frag(K, HEAD_DIM, 16, 32, lane);

  for (int kb = 0; kb < kend; kb += 32) {
    // V fragments: issue now, consumed after the softmax chain
    v16h vf[4];
#pragma unroll
    for (int j = 0; j < 4; ++j) vf[j] = load_frag(V, SEQP, j * 16, kb, lane);

    // scores: two 16x16 tiles over 32 k-columns, d = 64 (2 WMMAs each)
    v8f s0, s1;
#pragma unroll
    for (int r = 0; r < 8; ++r) { s0[r] = 0.f; s1[r] = 0.f; }
    s0 = wmma_f16(aq0, kf0, s0);
    s1 = wmma_f16(aq0, kf1, s1);
    s0 = wmma_f16(aq1, kf2, s0);
    s1 = wmma_f16(aq1, kf3, s1);

    // prefetch next k-block's K fragments; latency hides behind softmax + LDS
    if (kb + 32 < kend) {
      kf0 = load_frag(K, HEAD_DIM, kb + 32, 0, lane);
      kf1 = load_frag(K, HEAD_DIM, kb + 48, 0, lane);
      kf2 = load_frag(K, HEAD_DIM, kb + 32, 32, lane);
      kf3 = load_frag(K, HEAD_DIM, kb + 48, 32, lane);
    }

    // scale + causal mask + block row-max
    int kc0 = kb + (lane & 15);
    int kc1 = kc0 + 16;
    float bmax[8];
#pragma unroll
    for (int r = 0; r < 8; ++r) {
      int qrow = q0 + r + rowhalf;
      float v0 = (kc0 <= qrow) ? s0[r] * 0.125f : -1e30f;
      float v1 = (kc1 <= qrow) ? s1[r] * 0.125f : -1e30f;
      s0[r] = v0; s1[r] = v1;
      bmax[r] = rmax16(fmaxf(v0, v1));
    }

    // online softmax update
    float psum[8];
#pragma unroll
    for (int r = 0; r < 8; ++r) {
      float Mn  = fmaxf(mrow[r], bmax[r]);
      float fac = __expf(mrow[r] - Mn);
      float e0  = __expf(s0[r] - Mn);
      float e1  = __expf(s1[r] - Mn);
      s0[r] = e0; s1[r] = e1;
      psum[r] = rsum16(e0 + e1);
      mrow[r] = Mn;
      lrow[r] = lrow[r] * fac + psum[r];
#pragma unroll
      for (int j = 0; j < 4; ++j) o[j][r] *= fac;
    }

    // P: C layout -> A layout through per-wave LDS tile (wave-private slice)
#pragma unroll
    for (int r = 0; r < 8; ++r) {
      lds[wave][r + rowhalf][lane & 15]        = (_Float16)s0[r];
      lds[wave][r + rowhalf][16 + (lane & 15)] = (_Float16)s1[r];
    }
    asm volatile("s_wait_dscnt 0x0" ::: "memory");  // CDNA5 split DS counter wait
    v16h pa = lds_frag(&lds[wave][lane & 15][(lane >> 4) << 3]);

    // PV: O[16x64] += P[16x32] @ V[32x64]; columns of B contiguous in vT
#pragma unroll
    for (int j = 0; j < 4; ++j) o[j] = wmma_f16(pa, vf[j], o[j]);
  }

  // normalize + scatter into hbuf[B,S,1024] (head-interleaved, f16)
#pragma unroll
  for (int j = 0; j < 4; ++j) {
    int dcol = h * HEAD_DIM + j * 16 + (lane & 15);
#pragma unroll
    for (int r = 0; r < 8; ++r) {
      int qrow = q0 + r + rowhalf;
      float val = o[j][r] / lrow[r];
      hbuf[((long)b * SEQ + qrow) * D_MODEL + dcol] = (_Float16)val;
    }
  }
}

// ---------------------------------------------------------------------------
// host launch
// ---------------------------------------------------------------------------
extern "C" void kernel_launch(void* const* d_in, const int* in_sizes, int n_in,
                              void* d_out, int out_size, void* d_ws, size_t ws_size,
                              hipStream_t stream) {
  const float* x         = (const float*)d_in[0];
  const int*   positions = (const int*)d_in[1];
  const float* Wqkv      = (const float*)d_in[2];
  const float* bqkv      = (const float*)d_in[3];
  const float* Wo        = (const float*)d_in[4];
  const float* bo        = (const float*)d_in[5];
  float* out = (float*)d_out;

  const long M = (long)BATCH * SEQ;  // 8192

  char* ws = (char*)d_ws;
  auto carve = [&](size_t bytes) {
    char* p = ws;
    ws += (bytes + 255) & ~(size_t)255;
    return p;
  };
  _Float16* xh    = (_Float16*)carve(M * D_MODEL * 2);                         // 16 MB
  _Float16* wqkvh = (_Float16*)carve((size_t)3 * D_MODEL * D_MODEL * 2);       //  6 MB
  _Float16* woh   = (_Float16*)carve((size_t)D_MODEL * D_MODEL * 2);           //  2 MB
  _Float16* qkvh  = (_Float16*)carve(M * 3 * D_MODEL * 2);                     // 48 MB
  _Float16* qh    = (_Float16*)carve((size_t)BATCH * N_HEADS * SEQ  * HEAD_DIM * 2);
  _Float16* kh    = (_Float16*)carve((size_t)BATCH * N_HEADS * SEQP * HEAD_DIM * 2);
  _Float16* vT    = (_Float16*)carve((size_t)BATCH * N_HEADS * HEAD_DIM * SEQP * 2);
  _Float16* hbuf  = (_Float16*)carve(M * D_MODEL * 2);                         // 16 MB

  // 1) converts
  cvt_f32_f16<<<2048, 256, 0, stream>>>(x, xh, M * D_MODEL);
  cvt_f32_f16<<<2048, 256, 0, stream>>>(Wqkv, wqkvh, (long)3 * D_MODEL * D_MODEL);
  cvt_f32_f16<<<1024, 256, 0, stream>>>(Wo, woh, (long)D_MODEL * D_MODEL);

  // 2) QKV projection: [8192,1024] @ [1024,3072] -> f16
  gemm_wmma<<<dim3(3 * D_MODEL / 64, M / 256), 256, 0, stream>>>(
      xh, wqkvh, bqkv, nullptr, qkvh, (int)M, 3 * D_MODEL, D_MODEL);

  // 3) RoPE + per-head packing (+ V transpose, zero padding)
  {
    long threads = (long)BATCH * N_HEADS * SEQP * 32;
    rope_pack<<<(int)((threads + 255) / 256), 256, 0, stream>>>(
        qkvh, positions, qh, kh, vT);
  }

  // 4) causal flash attention
  attn_flash<<<dim3(BATCH * N_HEADS, SEQ / 128), 256, 0, stream>>>(qh, kh, vT, hbuf);

  // 5) output projection: [8192,1024] @ [1024,1024] + bo -> f32 d_out
  gemm_wmma<<<dim3(D_MODEL / 64, M / 256), 256, 0, stream>>>(
      hbuf, woh, bo, out, nullptr, (int)M, D_MODEL, D_MODEL);
}
